// DeepEMD_19524921327621
// MI455X (gfx1250) — matrix operand order repaired and checked by reading, not compiled, not run on hardware
//
#include <hip/hip_runtime.h>

// ---------------------------------------------------------------------------
// DeepEMD for MI455X (gfx1250): bf16 WMMA sim-GEMM + LDS-resident Sinkhorn.
// proto: [1,20,640,8,8] f32, query: [600,640,8,8] f32 -> logits [600,20] f32
//
// All Sinkhorn math is kept in the log2 domain (pre-scaled by log2(e)) so the
// inner loops map 1:1 onto v_exp_f32 / v_log_f32 (which are base-2 on AMD HW),
// avoiding one v_mul per exponential across ~2e9 exponentials.
// ---------------------------------------------------------------------------

typedef __attribute__((ext_vector_type(16))) __bf16        v16bf;
typedef __attribute__((ext_vector_type(8)))  float         v8f;
typedef __attribute__((ext_vector_type(8)))  unsigned int  v8u;

#define WAY    20
#define NQ     600
#define NC     640
#define NPOS   64          // 8*8 spatial positions
#define KSTEPS 20          // 640 / 32 (bf16 WMMA K=32)
#define FRAG_PER_IMG (4 * KSTEPS * 32 * 16)   // 40960 bf16 per image
#define CBSTRIDE (KSTEPS * 512)               // 10240 ushorts per col-block

#define LOG2E     1.4426950408889634f
#define MC_SCALE  28.853900817779268f         // 20 * log2(e)   (eps = 0.05)
#define INV_MC    0.0346573590279973f         // 0.05 / log2(e) (mC2 -> sim-1)

__device__ __forceinline__ float fexp2(float x) { return __builtin_amdgcn_exp2f(x); }
__device__ __forceinline__ float flog2(float x) { return __builtin_amdgcn_logf(x); }

// round-to-nearest-even f32 -> bf16
__device__ __forceinline__ unsigned short f2bf(float x) {
  unsigned int u = __float_as_uint(x);
  u += 0x7FFFu + ((u >> 16) & 1u);
  return (unsigned short)(u >> 16);
}

// ---------------------------------------------------------------------------
// Kernel 1: per-channel spatial means (adaptive_avg_pool2d to 1x1)
// ---------------------------------------------------------------------------
__global__ __launch_bounds__(256) void means_kernel(
    const float* __restrict__ proto, const float* __restrict__ query,
    float* __restrict__ Bmean, float* __restrict__ Qmean) {
  const int NP = WAY * NC;       // 12800
  const int NQC = NQ * NC;       // 384000
  int idx = blockIdx.x * 256 + threadIdx.x;
  if (idx < NP) {
    const float* p = proto + (size_t)idx * NPOS;
    float s = 0.f;
    #pragma unroll 4
    for (int i = 0; i < NPOS; ++i) s += p[i];
    Bmean[idx] = s * (1.f / NPOS);
  } else if (idx < NP + NQC) {
    int j = idx - NP;
    const float* p = query + (size_t)j * NPOS;
    float s = 0.f;
    #pragma unroll 4
    for (int i = 0; i < NPOS; ++i) s += p[i];
    Qmean[j] = s * (1.f / NPOS);
  }
}

// ---------------------------------------------------------------------------
// Kernel 2: center, L2-normalize over C, bf16-convert, pre-swizzle into the
// CDNA5 WMMA fragment layouts (ISA 7.12.2):
//   isB==0 (A 16x32, 16-bit): K = (e<8?0:16) + 8*(lane>>4) + 2*((e>>1)&3) + (e&1)
//   isB==1 (B 32x16, 16-bit): N = lane%16, K = (e<8?0:16) + 8*(lane>>4) + (e&7)
// frag[img][rb(4)][ks(20)][lane(32)][e(16)]  (32B contiguous per lane)
// ---------------------------------------------------------------------------
__global__ __launch_bounds__(256) void frag_kernel(
    const float* __restrict__ src, unsigned short* __restrict__ frag, int isB) {
  __shared__ float meanC[NPOS];
  __shared__ float invn[NPOS];
  const int img = blockIdx.x;
  const int tid = threadIdx.x;
  const float* sb = src + (size_t)img * NC * NPOS;

  if (tid < NPOS) {
    float s = 0.f;
    for (int c = 0; c < NC; ++c) s += sb[c * NPOS + tid];
    meanC[tid] = s * (1.f / NC);
  }
  __syncthreads();
  if (tid < NPOS) {
    float m = meanC[tid], ss = 0.f;
    for (int c = 0; c < NC; ++c) { float d = sb[c * NPOS + tid] - m; ss += d * d; }
    invn[tid] = 1.f / fmaxf(sqrtf(ss), 1e-8f);
  }
  __syncthreads();

  for (int t = tid; t < FRAG_PER_IMG; t += 256) {
    int e    = t & 15;
    int lane = (t >> 4) & 31;
    int rem  = t >> 9;            // rb*20 + ks
    int ks   = rem % KSTEPS;
    int rb   = rem / KSTEPS;
    int half = lane >> 4;
    int row  = rb * 16 + (lane & 15);
    int k    = isB ? (((e < 8) ? 0 : 16) + 8 * half + (e & 7))
                   : (((e < 8) ? 0 : 16) + 8 * half + (((e >> 1) & 3) << 1) + (e & 1));
    int c    = ks * 32 + k;
    float v  = (sb[c * NPOS + row] - meanC[row]) * invn[row];
    frag[(size_t)img * FRAG_PER_IMG + t] = f2bf(v);
  }
}

// ---------------------------------------------------------------------------
// Kernel 3: EMD marginals per (q,w) pair; store log2 of rescaled marginals.
// ---------------------------------------------------------------------------
__global__ __launch_bounds__(64) void marg_kernel(
    const float* __restrict__ query, const float* __restrict__ proto,
    const float* __restrict__ Bmean, const float* __restrict__ Qmean,
    float* __restrict__ log_r, float* __restrict__ log_c) {
  __shared__ float Bw[NC];
  __shared__ float Qq[NC];
  __shared__ float red[64];
  const int bid = blockIdx.x;
  const int q = bid / WAY, w = bid % WAY;
  const int tid = threadIdx.x;

  for (int c = tid; c < NC; c += 64) {
    Bw[c] = Bmean[w * NC + c];
    Qq[c] = Qmean[q * NC + c];
  }
  __syncthreads();

  const float* qb = query + (size_t)q * NC * NPOS + tid;
  const float* pb = proto + (size_t)w * NC * NPOS + tid;
  float rs = 0.f, cs = 0.f;
  for (int c = 0; c < NC; ++c) {
    rs = fmaf(qb[c * NPOS], Bw[c], rs);
    cs = fmaf(pb[c * NPOS], Qq[c], cs);
  }
  rs = fmaxf(rs, 0.f) + 0.00101f;   // relu(x)+0.001 then relu(.)+1e-5
  cs = fmaxf(cs, 0.f) + 0.00101f;

  red[tid] = rs; __syncthreads();
  for (int s = 32; s > 0; s >>= 1) { if (tid < s) red[tid] += red[tid + s]; __syncthreads(); }
  float sumR = red[0]; __syncthreads();
  red[tid] = cs; __syncthreads();
  for (int s = 32; s > 0; s >>= 1) { if (tid < s) red[tid] += red[tid + s]; __syncthreads(); }
  float sumC = red[0];

  // log2 of scaled marginals (Sinkhorn runs in log2 domain)
  log_r[(size_t)bid * NPOS + tid] = flog2(rs * (64.f / sumR));
  log_c[(size_t)bid * NPOS + tid] = flog2(cs * (64.f / sumC));
}

// ---------------------------------------------------------------------------
// Kernel 4: per (q,w) pair: 64x64 sim via v_wmma_f32_16x16x32_bf16
// (register double-buffered K loop), then 20 log2-domain Sinkhorn iterations
// in LDS with all 128 threads (2 threads per row/col), then logits.
//   mC2 = 20*(sim-1)*log2(e);  fs = 20*f*log2(e);  gs = 20*g*log2(e)
//   flow = exp2(mC2 + fs + gs); sim = 1 + INV_MC*mC2
// ---------------------------------------------------------------------------
__global__ __launch_bounds__(128) void emd_kernel(
    const unsigned short* __restrict__ qfrag, const unsigned short* __restrict__ pfrag,
    const float* __restrict__ log_r, const float* __restrict__ log_c,
    float* __restrict__ out) {
  __shared__ float mC[64 * 65];   // stride 65: conflict-free row & col passes
  __shared__ float fs[64];
  __shared__ float gs[64];
  __shared__ float pm[128];
  __shared__ float ps[128];

  const int bid  = blockIdx.x;
  const int q    = bid / WAY, w = bid % WAY;
  const int tid  = threadIdx.x;
  const int wave = tid >> 5, lane = tid & 31;
  const int half = lane >> 4, ln = lane & 15;

  // ---- sim GEMM: wave r computes rows [16r,16r+16) x cols [0,64) ----
  v8f a0 = {}, a1 = {}, a2 = {}, a3 = {};
  const unsigned short* aB = qfrag + (size_t)(q * 4 + wave) * CBSTRIDE + lane * 16;
  const unsigned short* bB = pfrag + (size_t)(w * 4) * CBSTRIDE + lane * 16;

  // prologue: fragments for ks = 0
  v8u Ac  = *(const v8u*)(aB);
  v8u B0c = *(const v8u*)(bB);
  v8u B1c = *(const v8u*)(bB + 1 * CBSTRIDE);
  v8u B2c = *(const v8u*)(bB + 2 * CBSTRIDE);
  v8u B3c = *(const v8u*)(bB + 3 * CBSTRIDE);

  #pragma unroll 2
  for (int ks = 0; ks < KSTEPS; ++ks) {
    // issue next-iteration loads before consuming current fragments
    int kn = ks + 1; if (kn == KSTEPS) kn = 0;   // last prefetch is dead
    const unsigned short* an = aB + kn * 512;
    const unsigned short* bn = bB + kn * 512;
    v8u An  = *(const v8u*)(an);
    v8u B0n = *(const v8u*)(bn);
    v8u B1n = *(const v8u*)(bn + 1 * CBSTRIDE);
    v8u B2n = *(const v8u*)(bn + 2 * CBSTRIDE);
    v8u B3n = *(const v8u*)(bn + 3 * CBSTRIDE);

    v16bf av = __builtin_bit_cast(v16bf, Ac);
    a0 = __builtin_amdgcn_wmma_f32_16x16x32_bf16(false, av, false,
           __builtin_bit_cast(v16bf, B0c), (short)0, a0, false, false);
    a1 = __builtin_amdgcn_wmma_f32_16x16x32_bf16(false, av, false,
           __builtin_bit_cast(v16bf, B1c), (short)0, a1, false, false);
    a2 = __builtin_amdgcn_wmma_f32_16x16x32_bf16(false, av, false,
           __builtin_bit_cast(v16bf, B2c), (short)0, a2, false, false);
    a3 = __builtin_amdgcn_wmma_f32_16x16x32_bf16(false, av, false,
           __builtin_bit_cast(v16bf, B3c), (short)0, a3, false, false);

    Ac = An; B0c = B0n; B1c = B1n; B2c = B2n; B3c = B3n;
  }

  // C/D layout: VGPR v -> M = v + 8*half, N = lane%16.  Store mC2.
  #pragma unroll
  for (int v = 0; v < 8; ++v) {
    int r = wave * 16 + v + 8 * half;
    float* mr = &mC[r * 65 + ln];
    mr[0]  = MC_SCALE * (a0[v] - 1.f);
    mr[16] = MC_SCALE * (a1[v] - 1.f);
    mr[32] = MC_SCALE * (a2[v] - 1.f);
    mr[48] = MC_SCALE * (a3[v] - 1.f);
  }

  float lr2 = 0.f, lc2 = 0.f;
  if (tid < 64) {
    lr2 = log_r[(size_t)bid * 64 + tid];
    lc2 = log_c[(size_t)bid * 64 + tid];
    fs[tid] = 0.f;
    gs[tid] = 0.f;
  }
  __syncthreads();

  // ---- Sinkhorn: 2 threads per row/col, each over 32 elements ----
  const int ri  = tid & 63;     // row (f-pass) / col (g-pass) index
  const int off = (tid >> 6) * 32;
  const float* rowp = &mC[ri * 65 + off];

  for (int it = 0; it < 20; ++it) {
    // f partials: LSE over columns
    {
      float m = -1e30f;
      #pragma unroll 8
      for (int j = 0; j < 32; ++j) m = fmaxf(m, rowp[j] + gs[off + j]);
      float s = 0.f;
      #pragma unroll 8
      for (int j = 0; j < 32; ++j) s += fexp2(rowp[j] + gs[off + j] - m);
      pm[tid] = m; ps[tid] = s;
    }
    __syncthreads();
    if (tid < 64) {
      float m0 = pm[tid], m1 = pm[tid + 64];
      float M = fmaxf(m0, m1);
      float S = ps[tid] * fexp2(m0 - M) + ps[tid + 64] * fexp2(m1 - M);
      fs[tid] = lr2 - (M + flog2(S));
    }
    __syncthreads();
    // g partials: LSE over rows
    {
      float m = -1e30f;
      #pragma unroll 8
      for (int i = 0; i < 32; ++i) m = fmaxf(m, mC[(off + i) * 65 + ri] + fs[off + i]);
      float s = 0.f;
      #pragma unroll 8
      for (int i = 0; i < 32; ++i) s += fexp2(mC[(off + i) * 65 + ri] + fs[off + i] - m);
      pm[tid] = m; ps[tid] = s;
    }
    __syncthreads();
    if (tid < 64) {
      float m0 = pm[tid], m1 = pm[tid + 64];
      float M = fmaxf(m0, m1);
      float S = ps[tid] * fexp2(m0 - M) + ps[tid + 64] * fexp2(m1 - M);
      gs[tid] = lc2 - (M + flog2(S));
    }
    __syncthreads();
  }

  // ---- logits = sum(sim * flow) * (T/n) ----
  float part = 0.f;
  const int base = tid * 32;
  #pragma unroll 4
  for (int t = 0; t < 32; ++t) {
    int idx = base + t;
    int i = idx >> 6, j = idx & 63;
    float m = mC[i * 65 + j];
    part += (1.f + INV_MC * m) * fexp2(m + fs[i] + gs[j]);
  }
  __syncthreads();
  ps[tid] = part;
  __syncthreads();
  for (int s = 64; s > 0; s >>= 1) { if (tid < s) ps[tid] += ps[tid + s]; __syncthreads(); }
  if (tid == 0) out[bid] = ps[0] * (12.5f / 64.f);
}

// ---------------------------------------------------------------------------
// Host launcher. Workspace layout (all offsets 256B-aligned, ~58.6 MB total):
//   Bmean   [20,640]  f32 @ 0          (51200 B)
//   Qmean   [600,640] f32 @ 51200      (1536000 B)
//   log_r   [12000,64]f32 @ 1587200    (3072000 B)   (log2 domain)
//   log_c   [12000,64]f32 @ 4659200    (3072000 B)   (log2 domain)
//   pfrag   bf16 B-frags  @ 7731200    (1638400 B)
//   qfrag   bf16 A-frags  @ 9369600    (49152000 B)
// ---------------------------------------------------------------------------
extern "C" void kernel_launch(void* const* d_in, const int* in_sizes, int n_in,
                              void* d_out, int out_size, void* d_ws, size_t ws_size,
                              hipStream_t stream) {
  const float* proto = (const float*)d_in[0];   // [1,20,640,8,8]
  const float* query = (const float*)d_in[1];   // [600,640,8,8]
  char* ws = (char*)d_ws;
  float*          Bmean = (float*)(ws + 0);
  float*          Qmean = (float*)(ws + 51200);
  float*          log_r = (float*)(ws + 1587200);
  float*          log_c = (float*)(ws + 4659200);
  unsigned short* pfrag = (unsigned short*)(ws + 7731200);
  unsigned short* qfrag = (unsigned short*)(ws + 9369600);

  means_kernel<<<(WAY * NC + NQ * NC + 255) / 256, 256, 0, stream>>>(proto, query, Bmean, Qmean);
  frag_kernel<<<WAY, 256, 0, stream>>>(proto, pfrag, 1);
  frag_kernel<<<NQ, 256, 0, stream>>>(query, qfrag, 0);
  marg_kernel<<<NQ * WAY, 64, 0, stream>>>(query, proto, Bmean, Qmean, log_r, log_c);
  emd_kernel<<<NQ * WAY, 128, 0, stream>>>(qfrag, pfrag, log_r, log_c, (float*)d_out);
}
